// LMMLayer_1005022347953
// MI455X (gfx1250) — matrix operand order repaired and checked by reading, not compiled, not run on hardware
//
#include <hip/hip_runtime.h>

#define B_ 64
#define S_ 256
#define IN_ 512
#define H_ 256
#define INNER_ 512

typedef __bf16 bf16_t;
typedef __attribute__((ext_vector_type(16))) bf16_t v16bf;
typedef __attribute__((ext_vector_type(8)))  bf16_t v8bf;
typedef __attribute__((ext_vector_type(8)))  float  v8f;
typedef __attribute__((ext_vector_type(4)))  float  f32x4;
typedef __attribute__((ext_vector_type(4)))  unsigned int u32x4;
typedef __attribute__((ext_vector_type(4)))  int    i32x4;
typedef __attribute__((ext_vector_type(8)))  int    i32x8;

#if defined(__AMDGCN__) && __has_builtin(__builtin_amdgcn_tensor_load_to_lds)
#define HAVE_TDM 1
#else
#define HAVE_TDM 0
#endif

__device__ __forceinline__ v8f wmma_bf16(v16bf a, v16bf b, v8f c) {
  return __builtin_amdgcn_wmma_f32_16x16x32_bf16(false, a, false, b, (short)0, c,
                                                 false, false);
}

__device__ __forceinline__ v16bf cat8(v8bf lo, v8bf hi) {
  return __builtin_shufflevector(lo, hi, 0, 1, 2, 3, 4, 5, 6, 7,
                                 8, 9, 10, 11, 12, 13, 14, 15);
}

// A fragment, bf16 source, K-contiguous rows: element(m,k) = base[m*rs + k]
__device__ __forceinline__ v16bf fragA_bf(const bf16_t* base, long rs, int lane) {
  const bf16_t* p = base + (long)(lane & 15) * rs + ((lane >> 4) << 3);
  v8bf lo = *(const v8bf*)p;
  v8bf hi = *(const v8bf*)(p + 16);
  return cat8(lo, hi);
}

// B fragment from TRANSPOSED storage bt[n*ns + k]: element(k,n), K contiguous.
__device__ __forceinline__ v16bf fragB_T(const bf16_t* bt, long ns, int lane) {
  const bf16_t* p = bt + (long)(lane & 15) * ns + ((lane >> 4) << 3);
  v8bf lo = *(const v8bf*)p;
  v8bf hi = *(const v8bf*)(p + 16);
  return cat8(lo, hi);
}

__device__ __forceinline__ float sigmoidf_(float x) { return 1.0f / (1.0f + __expf(-x)); }

#if HAVE_TDM
// TDM: DMA a 64-row x 32-col bf16 tile of a row-major [B_*S_][IN_] tensor
// into LDS at lds_off. Issued by one wave; completion via s_wait_tensorcnt.
__device__ __forceinline__ void tdm_load_tile(const bf16_t* gsrc, unsigned lds_off) {
  unsigned long long ga = (unsigned long long)(size_t)gsrc;
  u32x4 g0;
  g0[0] = 1u;                                   // count=1, user descriptor
  g0[1] = lds_off;                              // D#.lds_addr
  g0[2] = (unsigned)ga;                         // global_addr[31:0]
  g0[3] = (unsigned)((ga >> 32) & 0x01FFFFFFu) | (2u << 30);  // addr[56:32] | type=2
  i32x8 g1;
  g1[0] = (int)(1u << 16);                      // wg_mask=0, data_size=1 (2 bytes)
  g1[1] = (int)((unsigned)(IN_ & 0xFFFF) << 16);          // tensor_dim0[15:0]
  g1[2] = (int)((unsigned)(B_ * S_ & 0xFFFF) << 16);      // td0 hi=0 | tensor_dim1 lo
  g1[3] = (int)(((unsigned)(B_ * S_) >> 16) | (32u << 16)); // td1 hi | tile_dim0=32
  g1[4] = (int)64u;                             // tile_dim1=64, tile_dim2=0
  g1[5] = (int)IN_;                             // tensor_dim0_stride=512
  g1[6] = 0;
  g1[7] = 0;
  i32x4 z4 = {0, 0, 0, 0};
#if __clang_major__ >= 23
  i32x8 z8 = {0, 0, 0, 0, 0, 0, 0, 0};
  __builtin_amdgcn_tensor_load_to_lds(g0, g1, z4, z4, z8, 0);
#else
  __builtin_amdgcn_tensor_load_to_lds(g0, g1, z4, z4, 0);
#endif
}
#endif

// ---------------------------------------------------------------------------
// Init: transposed fp32 masters + bf16 shadows, zero momentum.
// ---------------------------------------------------------------------------
__global__ void lmm_init(const float* __restrict__ W1, const float* __restrict__ b1,
                         const float* __restrict__ W2, const float* __restrict__ b2,
                         float* __restrict__ W1fT, float* __restrict__ b1f,
                         float* __restrict__ W2fT, float* __restrict__ b2f,
                         float* __restrict__ S1T, float* __restrict__ Sb1,
                         float* __restrict__ S2T, float* __restrict__ Sb2,
                         unsigned short* __restrict__ W1hT_, unsigned short* __restrict__ W2hT_,
                         unsigned short* __restrict__ W2h_) {
  size_t i = (size_t)blockIdx.x * 256 + threadIdx.x;
  bf16_t* W1hT = (bf16_t*)W1hT_;
  bf16_t* W2hT = (bf16_t*)W2hT_;
  bf16_t* W2h  = (bf16_t*)W2h_;
  if (i < (size_t)H_ * INNER_) {
    {  // W1 (H_ x INNER_) -> transposed [INNER_][H_]
      size_t m = i / INNER_, n = i % INNER_;
      float w = W1[i];
      W1fT[n * H_ + m] = w;
      W1hT[n * H_ + m] = (bf16_t)w;
      S1T[i] = 0.0f;
    }
    {  // W2 (INNER_ x H_) -> transposed [H_][INNER_] + row-major shadow
      size_t r = i / H_, n = i % H_;
      float w = W2[i];
      W2fT[n * INNER_ + r] = w;
      W2hT[n * INNER_ + r] = (bf16_t)w;
      W2h[i] = (bf16_t)w;
      S2T[i] = 0.0f;
    }
  }
  if (i < INNER_) { b1f[i] = b1[i]; Sb1[i] = 0.0f; }
  if (i < H_)     { b2f[i] = b2[i]; Sb2[i] = 0.0f; }
}

// One-shot: projection weights (IN_ x H_ fp32) -> transposed bf16 [H_][IN_]
__global__ void lmm_wcvt(const float* __restrict__ Wk, const float* __restrict__ Wv,
                         const float* __restrict__ Wq,
                         unsigned short* __restrict__ WkT_, unsigned short* __restrict__ WvT_,
                         unsigned short* __restrict__ WqT_) {
  const float* W; bf16_t* WT;
  if (blockIdx.y == 0)      { W = Wk; WT = (bf16_t*)WkT_; }
  else if (blockIdx.y == 1) { W = Wv; WT = (bf16_t*)WvT_; }
  else                      { W = Wq; WT = (bf16_t*)WqT_; }
  size_t i = (size_t)blockIdx.x * 256 + threadIdx.x;  // over IN_*H_
  size_t rk = i / H_, n = i % H_;                     // coalesced read
  WT[n * IN_ + rk] = (bf16_t)W[i];
}

// One-shot: x (B*S x IN fp32) -> row-major bf16
__global__ void lmm_xcvt(const float* __restrict__ x, unsigned short* __restrict__ xh_) {
  bf16_t* xh = (bf16_t*)xh_;
  size_t i = ((size_t)blockIdx.x * 256 + threadIdx.x) * 8;
  f32x4 a0 = *(const f32x4*)(x + i);
  f32x4 a1 = *(const f32x4*)(x + i + 4);
  v8bf v;
#pragma unroll
  for (int j = 0; j < 4; ++j) { v[j] = (bf16_t)a0[j]; v[j + 4] = (bf16_t)a1[j]; }
  *(v8bf*)(xh + i) = v;
}

// ---------------------------------------------------------------------------
// Projections: C(16384x256) = X(16384x512) @ W + b.
// X tile (bf16) double-buffered in LDS via the Tensor Data Mover; B fragments
// straight from the L2-resident transposed bf16 weights. Also emits bf16
// copies of K and Q for the scan kernels.
// ---------------------------------------------------------------------------
__global__ void lmm_proj(const unsigned short* __restrict__ xh_,
                         const unsigned short* __restrict__ WkT_, const float* __restrict__ bk,
                         const unsigned short* __restrict__ WvT_, const float* __restrict__ bv,
                         const unsigned short* __restrict__ WqT_, const float* __restrict__ bq,
                         float* __restrict__ Kb, float* __restrict__ Vb, float* __restrict__ Qb,
                         unsigned short* __restrict__ Kh_, unsigned short* __restrict__ Vh_,
                         unsigned short* __restrict__ Qh_) {
  const bf16_t* xh = (const bf16_t*)xh_;
  const bf16_t* WT; const float* bias; float* out; bf16_t* outh;
  if (blockIdx.z == 0)      { WT = (const bf16_t*)WkT_; bias = bk; out = Kb; outh = (bf16_t*)Kh_; }
  else if (blockIdx.z == 1) { WT = (const bf16_t*)WvT_; bias = bv; out = Vb; outh = (bf16_t*)Vh_; }
  else                      { WT = (const bf16_t*)WqT_; bias = bq; out = Qb; outh = (bf16_t*)Qh_; }

  __shared__ alignas(16) bf16_t Xs[2][64 * 32];  // double-buffered [row][k]

  int m0 = blockIdx.y << 6, n0 = blockIdx.x << 6;
  int lane = threadIdx.x & 31, wave = threadIdx.x >> 5;
  int wr = wave >> 1, wc = wave & 1;
  const bf16_t* xrow = xh + (size_t)m0 * IN_;

#if HAVE_TDM
  unsigned lds0 = (unsigned)(unsigned long long)(size_t)&Xs[0][0];
  if (wave == 0) {
    tdm_load_tile(xrow, lds0);
    __builtin_amdgcn_s_wait_tensorcnt(0);
  }
#else
  {
    int sr = threadIdx.x >> 2, sc = (threadIdx.x & 3) << 3;
    *(v8bf*)&Xs[0][sr * 32 + sc] = *(const v8bf*)&xrow[(size_t)sr * IN_ + sc];
  }
#endif
  __syncthreads();

  v8f acc0 = {}, acc1 = {};
  int buf = 0;
  const bf16_t* bt0 = WT + (size_t)(n0 + wc * 32) * IN_;
  const bf16_t* bt1 = WT + (size_t)(n0 + wc * 32 + 16) * IN_;
  for (int k0 = 0; k0 < IN_; k0 += 32) {
    if (k0 + 32 < IN_) {
#if HAVE_TDM
      if (wave == 0) tdm_load_tile(xrow + k0 + 32, lds0 + (buf ? 0u : 4096u));
#else
      int sr = threadIdx.x >> 2, sc = (threadIdx.x & 3) << 3;
      *(v8bf*)&Xs[buf ^ 1][sr * 32 + sc] =
          *(const v8bf*)&xrow[(size_t)sr * IN_ + k0 + 32 + sc];
#endif
    }
    v16bf a  = fragA_bf(&Xs[buf][wr * 16 * 32], 32L, lane);
    v16bf b0 = fragB_T(bt0 + k0, (long)IN_, lane);
    v16bf b1 = fragB_T(bt1 + k0, (long)IN_, lane);
    acc0 = wmma_bf16(a, b0, acc0);
    acc1 = wmma_bf16(a, b1, acc1);
#if HAVE_TDM
    if (wave == 0) __builtin_amdgcn_s_wait_tensorcnt(0);
#endif
    __syncthreads();
    buf ^= 1;
  }
  int mo = (lane >> 4) << 3, nn = lane & 15;
#pragma unroll
  for (int r = 0; r < 8; ++r) {
    int m = m0 + wr * 16 + mo + r;
    int c = n0 + wc * 32 + nn;
    float v0 = acc0[r] + bias[c];
    float v1 = acc1[r] + bias[c + 16];
    out[(size_t)m * H_ + c]      = v0;
    out[(size_t)m * H_ + c + 16] = v1;
    outh[(size_t)m * H_ + c]      = (bf16_t)v0;
    outh[(size_t)m * H_ + c + 16] = (bf16_t)v1;
  }
}

// ---------------------------------------------------------------------------
// Gates: theta/eta/alpha per time step
// ---------------------------------------------------------------------------
__global__ void lmm_gates(const float* __restrict__ x,
                          const float* __restrict__ Wth, const float* __restrict__ bth,
                          const float* __restrict__ Weta, const float* __restrict__ beta,
                          const float* __restrict__ Wal, const float* __restrict__ bal,
                          float* __restrict__ th, float* __restrict__ et, float* __restrict__ al) {
  int s = blockIdx.x;
  int b = threadIdx.x;  // 64 threads = one batch row each
  const f32x4* xr = (const f32x4*)(x + ((size_t)b * S_ + s) * IN_);
  const f32x4* w0 = (const f32x4*)Wth;
  const f32x4* w1 = (const f32x4*)Weta;
  const f32x4* w2 = (const f32x4*)Wal;
  float d0 = 0.0f, d1 = 0.0f, d2 = 0.0f;
  for (int i = 0; i < IN_ / 4; ++i) {
    f32x4 xv = xr[i], a = w0[i], c = w1[i], e = w2[i];
#pragma unroll
    for (int j = 0; j < 4; ++j) { d0 += xv[j] * a[j]; d1 += xv[j] * c[j]; d2 += xv[j] * e[j]; }
  }
  d0 += bth[0]; d1 += beta[0]; d2 += bal[0];
  float sp = d0 > 20.0f ? d0 : __logf(1.0f + __expf(d0));
  __shared__ float r0[64], r1[64], r2[64];
  r0[b] = sp; r1[b] = sigmoidf_(d1); r2[b] = sigmoidf_(d2);
  __syncthreads();
  for (int off = 32; off > 0; off >>= 1) {
    if (b < off) { r0[b] += r0[b + off]; r1[b] += r1[b + off]; r2[b] += r2[b + off]; }
    __syncthreads();
  }
  if (b == 0) {
    float m0 = r0[0] * (1.0f / 64.0f);
    th[s] = m0 < 1.0f ? m0 : 1.0f;
    et[s] = r1[0] * (1.0f / 64.0f);
    al[s] = r2[0] * (1.0f / 64.0f);
  }
}

// ---------------------------------------------------------------------------
// Stage A: fwd + bwd activations for step t (all-bf16 operand paths).
// ---------------------------------------------------------------------------
__global__ void lmm_stepA(const unsigned short* __restrict__ Kh_, const float* __restrict__ Vb,
                          const unsigned short* __restrict__ W1hT_, const float* __restrict__ b1f,
                          const unsigned short* __restrict__ W2hT_,
                          const unsigned short* __restrict__ W2h_, const float* __restrict__ b2f,
                          unsigned short* __restrict__ zT_, unsigned short* __restrict__ hT_,
                          unsigned short* __restrict__ dyT_, unsigned short* __restrict__ dhT_,
                          int t) {
  const bf16_t* Kh   = (const bf16_t*)Kh_;
  const bf16_t* W1hT = (const bf16_t*)W1hT_;
  const bf16_t* W2hT = (const bf16_t*)W2hT_;
  const bf16_t* W2h  = (const bf16_t*)W2h_;
  bf16_t* zT  = (bf16_t*)zT_;
  bf16_t* hT  = (bf16_t*)hT_;
  bf16_t* dyT = (bf16_t*)dyT_;
  bf16_t* dhT = (bf16_t*)dhT_;

  __shared__ alignas(16) float  hpre[16 * INNER_];
  __shared__ alignas(16) bf16_t hs[16 * INNER_];
  __shared__ alignas(16) bf16_t dys[16 * H_];

  int lane = threadIdx.x & 31, wave = threadIdx.x >> 5;
  int b0 = blockIdx.x << 4;
  int mo = (lane >> 4) << 3, nn = lane & 15;
  const bf16_t* zbase = Kh + ((size_t)b0 * S_ + t) * H_;

  // z^T [H_][B_], cols b0..b0+15 (bf16 v8bf reads)
  for (int i = threadIdx.x; i < 16 * (H_ / 8); i += 256) {
    int k = i / (H_ / 8);
    int mq = (i % (H_ / 8)) << 3;
    v8bf v = *(const v8bf*)&Kh[((size_t)(b0 + k) * S_ + t) * H_ + mq];
#pragma unroll
    for (int j = 0; j < 8; ++j) zT[(size_t)(mq + j) * B_ + b0 + k] = v[j];
  }

  // phase 1: h_pre = z @ W1 + b1 ; 4 column tiles per wave share each A frag
  {
    int j0w = wave << 6;
    v8f acc[4] = {{}, {}, {}, {}};
    for (int k0 = 0; k0 < H_; k0 += 32) {
      v16bf a = fragA_bf(zbase + k0, (long)S_ * H_, lane);
#pragma unroll
      for (int q = 0; q < 4; ++q) {
        v16bf b = fragB_T(W1hT + (size_t)(j0w + q * 16) * H_ + k0, (long)H_, lane);
        acc[q] = wmma_bf16(a, b, acc[q]);
      }
    }
#pragma unroll
    for (int q = 0; q < 4; ++q) {
      int col = j0w + q * 16 + nn;
      v8bf hv;
#pragma unroll
      for (int r = 0; r < 8; ++r) {
        int m = mo + r;
        float v = acc[q][r] + b1f[col];
        hpre[m * INNER_ + col] = v;
        bf16_t hb = (bf16_t)(v * sigmoidf_(v));
        hs[m * INNER_ + col] = hb;
        hv[r] = hb;
      }
      *(v8bf*)&hT[(size_t)col * B_ + b0 + mo] = hv;
    }
  }
  __syncthreads();

  // phase 2: y = h @ W2 + b2 ; dy = 2*(y - v)/(B*H)
  {
    int n0 = wave << 5;
    v8f acc0 = {}, acc1 = {};
    for (int k0 = 0; k0 < INNER_; k0 += 32) {
      v16bf a  = fragA_bf(hs + k0, (long)INNER_, lane);
      v16bf b0 = fragB_T(W2hT + (size_t)n0 * INNER_ + k0, (long)INNER_, lane);
      v16bf b1 = fragB_T(W2hT + (size_t)(n0 + 16) * INNER_ + k0, (long)INNER_, lane);
      acc0 = wmma_bf16(a, b0, acc0);
      acc1 = wmma_bf16(a, b1, acc1);
    }
#pragma unroll
    for (int q = 0; q < 2; ++q) {
      v8f acc = q ? acc1 : acc0;
      int n = n0 + q * 16 + nn;
      v8bf dv;
#pragma unroll
      for (int r = 0; r < 8; ++r) {
        int m = mo + r;
        float y = acc[r] + b2f[n];
        float d = y - Vb[((size_t)(b0 + m) * S_ + t) * H_ + n];
        bf16_t db = (bf16_t)(d * (2.0f / (float)(B_ * H_)));
        dys[m * H_ + n] = db;
        dv[r] = db;
      }
      *(v8bf*)&dyT[(size_t)n * B_ + b0 + mo] = dv;
    }
  }
  __syncthreads();

  // phase 3: dh = dy @ W2^T ; dh_pre = dh * silu'(h_pre)
  {
    int j0w = wave << 6;
    v8f acc[4] = {{}, {}, {}, {}};
    for (int k0 = 0; k0 < H_; k0 += 32) {
      v16bf a = fragA_bf(dys + k0, (long)H_, lane);
#pragma unroll
      for (int q = 0; q < 4; ++q) {
        v16bf b = fragB_T(W2h + (size_t)(j0w + q * 16) * H_ + k0, (long)H_, lane);
        acc[q] = wmma_bf16(a, b, acc[q]);
      }
    }
#pragma unroll
    for (int q = 0; q < 4; ++q) {
      int col = j0w + q * 16 + nn;
      v8bf dh;
#pragma unroll
      for (int r = 0; r < 8; ++r) {
        int m = mo + r;
        float xp = hpre[m * INNER_ + col];
        float sg = sigmoidf_(xp);
        float ds = sg * (1.0f + xp * (1.0f - sg));
        dh[r] = (bf16_t)(acc[q][r] * ds);
      }
      *(v8bf*)&dhT[(size_t)col * B_ + b0 + mo] = dh;
    }
  }
}

// ---------------------------------------------------------------------------
// Stage B: fused weight-gradient (K=64) + clip + momentum + decay update.
// blocks 0..63: W1; 64..127: W2; 128: b1; 129: b2.
// ---------------------------------------------------------------------------
__global__ void lmm_stepB(const unsigned short* __restrict__ zT_,
                          const unsigned short* __restrict__ hT_,
                          const unsigned short* __restrict__ dyT_,
                          const unsigned short* __restrict__ dhT_,
                          float* __restrict__ W1fT, float* __restrict__ S1T,
                          unsigned short* __restrict__ W1hT_,
                          float* __restrict__ W2fT, float* __restrict__ S2T,
                          unsigned short* __restrict__ W2hT_, unsigned short* __restrict__ W2h_,
                          float* __restrict__ b1f, float* __restrict__ Sb1,
                          float* __restrict__ b2f, float* __restrict__ Sb2,
                          const float* __restrict__ th, const float* __restrict__ et,
                          const float* __restrict__ al, int t) {
  const bf16_t* zT  = (const bf16_t*)zT_;
  const bf16_t* hT  = (const bf16_t*)hT_;
  const bf16_t* dyT = (const bf16_t*)dyT_;
  const bf16_t* dhT = (const bf16_t*)dhT_;
  float theta = th[t], eta = et[t], alpha = al[t];
  int lane = threadIdx.x & 31, wave = threadIdx.x >> 5;
  int mo = (lane >> 4) << 3, nn = lane & 15;
  int blk = blockIdx.x;

  if (blk < 64) {  // gW1[m,j] = sum_b z^T[m][b] * dh^T[j][b]
    int tile = blk * 8 + wave;
    int m0 = (tile >> 5) << 4, n0 = (tile & 31) << 4;
    v8f acc = {};
#pragma unroll
    for (int k0 = 0; k0 < B_; k0 += 32) {
      v16bf a = fragA_bf(zT + (size_t)m0 * B_ + k0, (long)B_, lane);
      v16bf b = fragB_T(dhT + (size_t)n0 * B_ + k0, (long)B_, lane);
      acc = wmma_bf16(a, b, acc);
    }
    size_t base = (size_t)(n0 + nn) * H_ + m0 + mo;
    f32x4 sl = *(f32x4*)(S1T + base), sh = *(f32x4*)(S1T + base + 4);
    f32x4 wl = *(f32x4*)(W1fT + base), wh = *(f32x4*)(W1fT + base + 4);
    v8bf wv;
#pragma unroll
    for (int r = 0; r < 8; ++r) {
      float g = fminf(fmaxf(acc[r], -1.0f), 1.0f);
      float sv = (r < 4 ? sl[r & 3] : sh[r & 3]);
      float wvv = (r < 4 ? wl[r & 3] : wh[r & 3]);
      float s = eta * sv - theta * g;
      float w = (1.0f - alpha) * wvv + s;
      if (r < 4) { sl[r & 3] = s; wl[r & 3] = w; } else { sh[r & 3] = s; wh[r & 3] = w; }
      wv[r] = (bf16_t)w;
    }
    *(f32x4*)(S1T + base) = sl;  *(f32x4*)(S1T + base + 4) = sh;
    *(f32x4*)(W1fT + base) = wl; *(f32x4*)(W1fT + base + 4) = wh;
    *(v8bf*)((bf16_t*)W1hT_ + base) = wv;
  } else if (blk < 128) {  // gW2[i,n] = sum_b h^T[i][b] * dy^T[n][b]
    int tile = (blk - 64) * 8 + wave;
    int i0 = (tile >> 4) << 4, n0 = (tile & 15) << 4;
    v8f acc = {};
#pragma unroll
    for (int k0 = 0; k0 < B_; k0 += 32) {
      v16bf a = fragA_bf(hT + (size_t)i0 * B_ + k0, (long)B_, lane);
      v16bf b = fragB_T(dyT + (size_t)n0 * B_ + k0, (long)B_, lane);
      acc = wmma_bf16(a, b, acc);
    }
    int n = n0 + nn;
    size_t base = (size_t)n * INNER_ + i0 + mo;
    f32x4 sl = *(f32x4*)(S2T + base), sh = *(f32x4*)(S2T + base + 4);
    f32x4 wl = *(f32x4*)(W2fT + base), wh = *(f32x4*)(W2fT + base + 4);
    v8bf wv;
    bf16_t* W2h = (bf16_t*)W2h_;
#pragma unroll
    for (int r = 0; r < 8; ++r) {
      float g = fminf(fmaxf(acc[r], -1.0f), 1.0f);
      float sv = (r < 4 ? sl[r & 3] : sh[r & 3]);
      float wvv = (r < 4 ? wl[r & 3] : wh[r & 3]);
      float s = eta * sv - theta * g;
      float w = (1.0f - alpha) * wvv + s;
      if (r < 4) { sl[r & 3] = s; wl[r & 3] = w; } else { sh[r & 3] = s; wh[r & 3] = w; }
      wv[r] = (bf16_t)w;
      W2h[(size_t)(i0 + mo + r) * H_ + n] = (bf16_t)w;
    }
    *(f32x4*)(S2T + base) = sl;  *(f32x4*)(S2T + base + 4) = sh;
    *(f32x4*)(W2fT + base) = wl; *(f32x4*)(W2fT + base + 4) = wh;
    *(v8bf*)((bf16_t*)W2hT_ + base) = wv;
  } else if (blk == 128) {  // b1
    for (int j = threadIdx.x; j < INNER_; j += 256) {
      float g = 0.0f;
      const v8bf* row = (const v8bf*)(dhT + (size_t)j * B_);
#pragma unroll
      for (int k8 = 0; k8 < B_ / 8; ++k8) {
        v8bf v = row[k8];
#pragma unroll
        for (int e = 0; e < 8; ++e) g += (float)v[e];
      }
      g = fminf(fmaxf(g, -1.0f), 1.0f);
      float s = eta * Sb1[j] - theta * g;
      Sb1[j] = s;
      b1f[j] = (1.0f - alpha) * b1f[j] + s;
    }
  } else {  // b2
    for (int j = threadIdx.x; j < H_; j += 256) {
      float g = 0.0f;
      const v8bf* row = (const v8bf*)(dyT + (size_t)j * B_);
#pragma unroll
      for (int k8 = 0; k8 < B_ / 8; ++k8) {
        v8bf v = row[k8];
#pragma unroll
        for (int e = 0; e < 8; ++e) g += (float)v[e];
      }
      g = fminf(fmaxf(g, -1.0f), 1.0f);
      float s = eta * Sb2[j] - theta * g;
      Sb2[j] = s;
      b2f[j] = (1.0f - alpha) * b2f[j] + s;
    }
  }
}

// ---------------------------------------------------------------------------
// Stage C: out_t = silu(q @ W1 + b1) @ W2 + b2 with UPDATED weights.
// ---------------------------------------------------------------------------
__global__ void lmm_stepC(const unsigned short* __restrict__ Qh_,
                          const unsigned short* __restrict__ W1hT_, const float* __restrict__ b1f,
                          const unsigned short* __restrict__ W2hT_, const float* __restrict__ b2f,
                          float* __restrict__ out, int t) {
  const bf16_t* Qh   = (const bf16_t*)Qh_;
  const bf16_t* W1hT = (const bf16_t*)W1hT_;
  const bf16_t* W2hT = (const bf16_t*)W2hT_;
  __shared__ alignas(16) bf16_t hs[16 * INNER_];
  int lane = threadIdx.x & 31, wave = threadIdx.x >> 5;
  int b0 = blockIdx.x << 4;
  int mo = (lane >> 4) << 3, nn = lane & 15;
  const bf16_t* qbase = Qh + ((size_t)b0 * S_ + t) * H_;

  {
    int j0w = wave << 6;
    v8f acc[4] = {{}, {}, {}, {}};
    for (int k0 = 0; k0 < H_; k0 += 32) {
      v16bf a = fragA_bf(qbase + k0, (long)S_ * H_, lane);
#pragma unroll
      for (int q = 0; q < 4; ++q) {
        v16bf b = fragB_T(W1hT + (size_t)(j0w + q * 16) * H_ + k0, (long)H_, lane);
        acc[q] = wmma_bf16(a, b, acc[q]);
      }
    }
#pragma unroll
    for (int q = 0; q < 4; ++q) {
      int col = j0w + q * 16 + nn;
#pragma unroll
      for (int r = 0; r < 8; ++r) {
        int m = mo + r;
        float v = acc[q][r] + b1f[col];
        hs[m * INNER_ + col] = (bf16_t)(v * sigmoidf_(v));
      }
    }
  }
  __syncthreads();

  {
    int n0 = wave << 5;
    v8f acc0 = {}, acc1 = {};
    for (int k0 = 0; k0 < INNER_; k0 += 32) {
      v16bf a  = fragA_bf(hs + k0, (long)INNER_, lane);
      v16bf b0 = fragB_T(W2hT + (size_t)n0 * INNER_ + k0, (long)INNER_, lane);
      v16bf b1 = fragB_T(W2hT + (size_t)(n0 + 16) * INNER_ + k0, (long)INNER_, lane);
      acc0 = wmma_bf16(a, b0, acc0);
      acc1 = wmma_bf16(a, b1, acc1);
    }
#pragma unroll
    for (int q = 0; q < 2; ++q) {
      v8f acc = q ? acc1 : acc0;
      int n = n0 + q * 16 + nn;
#pragma unroll
      for (int r = 0; r < 8; ++r) {
        int m = mo + r;
        float y = acc[r] + b2f[n];
        out[((size_t)(b0 + m) * S_ + t) * H_ + n] = y;
        if (t == S_ - 1)
          out[(size_t)B_ * S_ * H_ + (size_t)(b0 + m) * H_ + n] = y;
      }
    }
  }
}

// ---------------------------------------------------------------------------
extern "C" void kernel_launch(void* const* d_in, const int* in_sizes, int n_in,
                              void* d_out, int out_size, void* d_ws, size_t ws_size,
                              hipStream_t stream) {
  const float* x    = (const float*)d_in[0];
  const float* Wk   = (const float*)d_in[1];
  const float* bk   = (const float*)d_in[2];
  const float* Wv   = (const float*)d_in[3];
  const float* bv   = (const float*)d_in[4];
  const float* Wq   = (const float*)d_in[5];
  const float* bq   = (const float*)d_in[6];
  const float* Wth  = (const float*)d_in[7];
  const float* bth  = (const float*)d_in[8];
  const float* Weta = (const float*)d_in[9];
  const float* beta = (const float*)d_in[10];
  const float* Wal  = (const float*)d_in[11];
  const float* bal  = (const float*)d_in[12];
  const float* W1   = (const float*)d_in[13];
  const float* b1   = (const float*)d_in[14];
  const float* W2   = (const float*)d_in[15];
  const float* b2   = (const float*)d_in[16];
  float* out = (float*)d_out;

  float* ws = (float*)d_ws;
  size_t o = 0;
  float* W1fT = ws + o; o += (size_t)H_ * INNER_;
  float* b1f  = ws + o; o += INNER_;
  float* W2fT = ws + o; o += (size_t)INNER_ * H_;
  float* b2f  = ws + o; o += H_;
  float* S1T  = ws + o; o += (size_t)H_ * INNER_;
  float* Sb1  = ws + o; o += INNER_;
  float* S2T  = ws + o; o += (size_t)INNER_ * H_;
  float* Sb2  = ws + o; o += H_;
  float* Kb   = ws + o; o += (size_t)B_ * S_ * H_;
  float* Vb   = ws + o; o += (size_t)B_ * S_ * H_;
  float* Qb   = ws + o; o += (size_t)B_ * S_ * H_;
  float* th   = ws + o; o += S_;
  float* et   = ws + o; o += S_;
  float* al   = ws + o; o += S_;
  unsigned short* W1hT = (unsigned short*)(ws + o);
  unsigned short* W2hT = W1hT + (size_t)H_ * INNER_;
  unsigned short* W2h  = W2hT + (size_t)INNER_ * H_;
  unsigned short* zT   = W2h + (size_t)INNER_ * H_;
  unsigned short* hT   = zT + (size_t)H_ * B_;
  unsigned short* dyT  = hT + (size_t)INNER_ * B_;
  unsigned short* dhT  = dyT + (size_t)H_ * B_;
  unsigned short* WkT  = dhT + (size_t)INNER_ * B_;
  unsigned short* WvT  = WkT + (size_t)IN_ * H_;
  unsigned short* WqT  = WvT + (size_t)IN_ * H_;
  unsigned short* xhb  = WqT + (size_t)IN_ * H_;
  unsigned short* Kh   = xhb + (size_t)B_ * S_ * IN_;
  unsigned short* Vh   = Kh + (size_t)B_ * S_ * H_;
  unsigned short* Qh   = Vh + (size_t)B_ * S_ * H_;

  lmm_init<<<512, 256, 0, stream>>>(W1, b1, W2, b2, W1fT, b1f, W2fT, b2f,
                                    S1T, Sb1, S2T, Sb2, W1hT, W2hT, W2h);

  dim3 wg((IN_ * H_) / 256, 3);
  lmm_wcvt<<<wg, 256, 0, stream>>>(Wk, Wv, Wq, WkT, WvT, WqT);

  lmm_xcvt<<<(B_ * S_ * IN_) / (256 * 8), 256, 0, stream>>>(x, xhb);

  dim3 pg(H_ / 64, (B_ * S_) / 64, 3);
  lmm_proj<<<pg, 256, 0, stream>>>(xhb, WkT, bk, WvT, bv, WqT, bq,
                                   Kb, Vb, Qb, Kh, Vh, Qh);

  lmm_gates<<<S_, 64, 0, stream>>>(x, Wth, bth, Weta, beta, Wal, bal, th, et, al);

  for (int t = 0; t < S_; ++t) {
    lmm_stepA<<<B_ / 16, 256, 0, stream>>>(Kh, Vb, W1hT, b1f, W2hT, W2h, b2f,
                                           zT, hT, dyT, dhT, t);
    lmm_stepB<<<130, 256, 0, stream>>>(zT, hT, dyT, dhT,
                                       W1fT, S1T, W1hT, W2fT, S2T, W2hT, W2h,
                                       b1f, Sb1, b2f, Sb2, th, et, al, t);
    lmm_stepC<<<B_ / 16, 256, 0, stream>>>(Qh, W1hT, b1f, W2hT, b2f, out, t);
  }
}
